// HeadAttention_36670430773488
// MI455X (gfx1250) — matrix operand order repaired and checked by reading, not compiled, not run on hardware
//
#include <hip/hip_runtime.h>
#include <hip/hip_bf16.h>

typedef _Float16 half_t;
typedef __attribute__((ext_vector_type(16))) _Float16     v16h;
typedef __attribute__((ext_vector_type(8)))  _Float16     v8h;
typedef __attribute__((ext_vector_type(8)))  float        v8f;
typedef __attribute__((ext_vector_type(4)))  float        v4f;
typedef __attribute__((ext_vector_type(4)))  unsigned int v4u;
typedef __attribute__((ext_vector_type(8)))  int          v8i;
typedef __attribute__((ext_vector_type(4)))  int          v4i;

#define WMMA_F16(A, B, C) \
  __builtin_amdgcn_wmma_f32_16x16x32_f16(false, (A), false, (B), (short)0, (C), false, false)

__device__ __forceinline__ v16h cat16(v8h lo, v8h hi) {
  return __builtin_shufflevector(lo, hi, 0,1,2,3,4,5,6,7,8,9,10,11,12,13,14,15);
}

// ---------------------------------------------------------------------------
// TDM helper: DMA a [32 x 64] f16 tile (row stride 64 elements) from a
// [4096 x 64] f16 tensor in global memory into LDS.  Wave-level op.
// ---------------------------------------------------------------------------
__device__ __forceinline__ void tdm_load_tile(const half_t* gsrc,
                                              unsigned int lds_byte_off) {
  const unsigned long long ga = (unsigned long long)(uintptr_t)gsrc;
  v4u g0;
  g0[0] = 1u;                                   // count=1, user descriptor
  g0[1] = lds_byte_off;                         // lds_addr
  g0[2] = (unsigned int)(ga & 0xFFFFFFFFull);   // global_addr[31:0]
  g0[3] = (unsigned int)((ga >> 32) & 0x01FFFFFFu) | (2u << 30);  // [56:32] | type=2

  v8i g1;
  g1[0] = (int)(1u << 16);          // workgroup_mask=0, data_size=1 (2 bytes)
  g1[1] = (int)(64u << 16);         // tensor_dim0[15:0]=64 in bits[79:48]
  g1[2] = (int)(4096u << 16);       // tensor_dim0 hi=0 | tensor_dim1[15:0]=4096
  g1[3] = (int)(64u << 16);         // tensor_dim1 hi=0 | tile_dim0=64
  g1[4] = 32;                       // tile_dim1=32, tile_dim2=0
  g1[5] = 64;                       // tensor_dim0_stride low32 = 64
  g1[6] = 0;
  g1[7] = 0;

  v4i gz = {0, 0, 0, 0};
#if defined(__clang_major__) && (__clang_major__ >= 23)
  v8i gz8 = {0, 0, 0, 0, 0, 0, 0, 0};
  __builtin_amdgcn_tensor_load_to_lds(g0, g1, gz, gz, gz8, 0);
#else
  __builtin_amdgcn_tensor_load_to_lds(g0, g1, gz, gz, 0);
#endif
}

// ---------------------------------------------------------------------------
// Kernel 0: one-shot W f32 -> f16 conversion (3 x [64,512]).
// ---------------------------------------------------------------------------
__global__ __launch_bounds__(256) void wcvt(
    const float* __restrict__ Wq, const float* __restrict__ Wk,
    const float* __restrict__ Wv, half_t* __restrict__ W16)
{
  const int i = (blockIdx.x * 256 + threadIdx.x) * 4;
  const float* srcs[3] = {Wq, Wk, Wv};
  const int mat = i >> 15;
  const int off = i & 32767;
  const v4f f = *(const v4f*)(srcs[mat] + off);
  half_t* o = W16 + i;
  o[0] = (_Float16)f.x; o[1] = (_Float16)f.y;
  o[2] = (_Float16)f.z; o[3] = (_Float16)f.w;
}

// ---------------------------------------------------------------------------
// Kernel 1: fused QKV projection.  One wave per 16-row tile; N=64 per matrix
// (4 n-tiles), K=512 (16 chunks of 32).  Q pre-scaled by 1/sqrt(H).
// ---------------------------------------------------------------------------
__global__ __launch_bounds__(256) void qkv_proj(
    const float* __restrict__ x, const half_t* __restrict__ W16,
    half_t* __restrict__ Q, half_t* __restrict__ K, half_t* __restrict__ V)
{
  const int lane = threadIdx.x & 31;
  const int wave = threadIdx.x >> 5;
  const int lr   = lane & 15;
  const int hi   = lane >> 4;
  const int m0   = (blockIdx.x * 8 + wave) * 16;
  const int arow = m0 + lr;

  v8f acc[12] = {};

  for (int kc = 0; kc < 16; ++kc) {
    const float* xr = x + (size_t)arow * 512 + kc * 32 + 8 * hi;
    v16h a;
    #pragma unroll
    for (int e = 0; e < 8; ++e) a[e]     = (_Float16)xr[e];
    #pragma unroll
    for (int e = 0; e < 8; ++e) a[8 + e] = (_Float16)xr[16 + e];

    #pragma unroll
    for (int mat = 0; mat < 3; ++mat) {
      #pragma unroll
      for (int nt = 0; nt < 4; ++nt) {
        const half_t* wr = W16 + mat * 32768 + (size_t)(nt * 16 + lr) * 512
                                + kc * 32 + 8 * hi;
        v16h bv = cat16(*(const v8h*)wr, *(const v8h*)(wr + 16));
        acc[mat * 4 + nt] = WMMA_F16(a, bv, acc[mat * 4 + nt]);
      }
    }
  }

  const float qscale = 0.125f;        // 1/sqrt(64)
  half_t* Outs[3] = {Q, K, V};
  #pragma unroll
  for (int mat = 0; mat < 3; ++mat) {
    #pragma unroll
    for (int nt = 0; nt < 4; ++nt) {
      #pragma unroll
      for (int r = 0; r < 8; ++r) {
        float vv = acc[mat * 4 + nt][r];
        if (mat == 0) vv *= qscale;
        const int orow = m0 + r + 8 * hi;
        Outs[mat][(size_t)orow * 64 + nt * 16 + lr] = (half_t)vv;
      }
    }
  }
}

// ---------------------------------------------------------------------------
// Kernel 2: flash attention (causal), TDM double-buffered.  4 waves/block,
// 16 query rows per wave.  Wave 0 prefetches key-block jb+1 via the Tensor
// Data Mover while all waves compute on key-block jb.
// ---------------------------------------------------------------------------
__global__ __launch_bounds__(128) void attn(
    const half_t* __restrict__ Q, const half_t* __restrict__ K,
    const half_t* __restrict__ V, float* __restrict__ out)
{
  __shared__ half_t Kt[2][32 * 64];    // [buf][key row][h]  (TDM dest)  8 KB
  __shared__ half_t Vr[2][32 * 64];    // [buf][key row][h]  (TDM dest)  8 KB
  __shared__ half_t Vt[2][64 * 32];    // [buf][h][key row]  transposed  8 KB
  __shared__ half_t Pst[4][16 * 32];   // per-wave P staging             4 KB

  const int lane = threadIdx.x & 31;
  const int wave = threadIdx.x >> 5;
  const int lr   = lane & 15;
  const int hi   = lane >> 4;

  const int nIb   = 4096 / 64;
  const int b     = blockIdx.x / nIb;
  const int i0    = (blockIdx.x % nIb) * 64;
  const int rbase = i0 + wave * 16;

  const half_t* Qb = Q + (size_t)b * 4096 * 64;
  const half_t* Kb = K + (size_t)b * 4096 * 64;
  const half_t* Vb = V + (size_t)b * 4096 * 64;

  // Preload Q A-operand tiles (h 0..31 and 32..63).
  v16h aq[2];
  {
    const half_t* qr = Qb + (size_t)(rbase + lr) * 64;
    #pragma unroll
    for (int kc = 0; kc < 2; ++kc) {
      v8h lo = *(const v8h*)(qr + kc * 32 + 8 * hi);
      v8h hh = *(const v8h*)(qr + kc * 32 + 16 + 8 * hi);
      aq[kc] = cat16(lo, hh);
    }
  }

  v8f  o[4] = {};
  float mrow[8], lrow[8];
  #pragma unroll
  for (int r = 0; r < 8; ++r) { mrow[r] = -__builtin_huge_valf(); lrow[r] = 0.0f; }

  const unsigned int ldsK[2] = {(unsigned int)(uintptr_t)(void*)&Kt[0][0],
                                (unsigned int)(uintptr_t)(void*)&Kt[1][0]};
  const unsigned int ldsV[2] = {(unsigned int)(uintptr_t)(void*)&Vr[0][0],
                                (unsigned int)(uintptr_t)(void*)&Vr[1][0]};

  const int nj = i0 / 32 + 2;          // key blocks up to the diagonal (>= 2)

  // Prologue: kick off tile 0 into buffer 0.
  if (wave == 0) {
    tdm_load_tile(Kb, ldsK[0]);
    tdm_load_tile(Vb, ldsV[0]);
  }

  for (int jb = 0; jb < nj; ++jb) {
    const int s0 = jb * 32;
    const int p  = jb & 1;

    // All consumers of buffer 1-p (iteration jb-1) finished.
    __syncthreads();
    if (wave == 0) {
      if (jb + 1 < nj) {
        // Prefetch next tile into the other buffer, then wait until only
        // those two DMAs remain outstanding -> tile jb has landed.
        const size_t s1 = (size_t)(s0 + 32) * 64;
        tdm_load_tile(Kb + s1, ldsK[1 - p]);
        tdm_load_tile(Vb + s1, ldsV[1 - p]);
        __builtin_amdgcn_s_wait_tensorcnt(2);
      } else {
        __builtin_amdgcn_s_wait_tensorcnt(0);
      }
    }
    __syncthreads();                   // Kt[p] / Vr[p] visible to all waves

    // In-LDS transpose of V: Vr[p][s][h] -> Vt[p][h][s].
    {
      const int r = threadIdx.x >> 2;
      const int c = (threadIdx.x & 3) * 16;
      v8h a0 = *(const v8h*)&Vr[p][r * 64 + c];
      v8h a1 = *(const v8h*)&Vr[p][r * 64 + c + 8];
      #pragma unroll
      for (int e = 0; e < 8; ++e) Vt[p][(c + e) * 32 + r]     = a0[e];
      #pragma unroll
      for (int e = 0; e < 8; ++e) Vt[p][(c + 8 + e) * 32 + r] = a1[e];
    }
    __syncthreads();                   // Vt[p] ready

    // ---- S = Q @ K^T -------------------------------------------------------
    v8f s_acc[2];
    #pragma unroll
    for (int nt = 0; nt < 2; ++nt) {
      v8f sa = {};
      #pragma unroll
      for (int kc = 0; kc < 2; ++kc) {
        const half_t* kp = &Kt[p][(nt * 16 + lr) * 64 + kc * 32 + 8 * hi];
        v16h bv = cat16(*(const v8h*)kp, *(const v8h*)(kp + 16));
        sa = WMMA_F16(aq[kc], bv, sa);
      }
      s_acc[nt] = sa;
    }

    // ---- causal mask -------------------------------------------------------
    if (s0 + 31 > rbase) {
      #pragma unroll
      for (int nt = 0; nt < 2; ++nt) {
        const int col = s0 + nt * 16 + lr;
        #pragma unroll
        for (int r = 0; r < 8; ++r) {
          const int rowg = rbase + r + 8 * hi;
          if (col > rowg) s_acc[nt][r] = -__builtin_huge_valf();
        }
      }
    }

    // ---- online softmax (rows live in 16-lane halves) ----------------------
    float pm[8];
    #pragma unroll
    for (int r = 0; r < 8; ++r) pm[r] = fmaxf(s_acc[0][r], s_acc[1][r]);
    #pragma unroll
    for (int k = 0; k < 4; ++k) {
      const int off = 1 << k;
      #pragma unroll
      for (int r = 0; r < 8; ++r) pm[r] = fmaxf(pm[r], __shfl_xor(pm[r], off, 32));
    }

    float alpha[8];
    #pragma unroll
    for (int r = 0; r < 8; ++r) {
      const float mn = fmaxf(mrow[r], pm[r]);
      alpha[r] = __expf(mrow[r] - mn);
      mrow[r]  = mn;
    }

    float rs[8];
    #pragma unroll
    for (int r = 0; r < 8; ++r) {
      const float p0 = __expf(s_acc[0][r] - mrow[r]);
      const float p1 = __expf(s_acc[1][r] - mrow[r]);
      s_acc[0][r] = p0; s_acc[1][r] = p1;
      rs[r] = p0 + p1;
    }
    #pragma unroll
    for (int k = 0; k < 4; ++k) {
      const int off = 1 << k;
      #pragma unroll
      for (int r = 0; r < 8; ++r) rs[r] += __shfl_xor(rs[r], off, 32);
    }
    #pragma unroll
    for (int r = 0; r < 8; ++r) lrow[r] = lrow[r] * alpha[r] + rs[r];
    #pragma unroll
    for (int nt = 0; nt < 4; ++nt)
      #pragma unroll
      for (int r = 0; r < 8; ++r) o[nt][r] *= alpha[r];

    // ---- P: C-layout -> A-layout via per-wave LDS staging ------------------
    half_t* ps = Pst[wave];
    #pragma unroll
    for (int r = 0; r < 8; ++r) {
      const int rw = r + 8 * hi;
      ps[rw * 32 + lr]      = (half_t)s_acc[0][r];
      ps[rw * 32 + 16 + lr] = (half_t)s_acc[1][r];
    }
    asm volatile("s_wait_dscnt 0" ::: "memory");   // cross-lane LDS RAW

    v16h ap;
    {
      const half_t* pr = ps + lr * 32 + 8 * hi;
      ap = cat16(*(const v8h*)pr, *(const v8h*)(pr + 16));
    }

    // ---- O += P @ V --------------------------------------------------------
    #pragma unroll
    for (int nt = 0; nt < 4; ++nt) {
      const half_t* vp = &Vt[p][(nt * 16 + lr) * 32 + 8 * hi];
      v16h bv = cat16(*(const v8h*)vp, *(const v8h*)(vp + 16));
      o[nt] = WMMA_F16(ap, bv, o[nt]);
    }
  }

  // ---- epilogue ------------------------------------------------------------
  float* ob = out + (size_t)b * 4096 * 64;
  float inv[8];
  #pragma unroll
  for (int r = 0; r < 8; ++r) inv[r] = 1.0f / lrow[r];
  #pragma unroll
  for (int nt = 0; nt < 4; ++nt) {
    #pragma unroll
    for (int r = 0; r < 8; ++r) {
      const int rowg = rbase + r + 8 * hi;
      ob[(size_t)rowg * 64 + nt * 16 + lr] = o[nt][r] * inv[r];
    }
  }
}

// ---------------------------------------------------------------------------
extern "C" void kernel_launch(void* const* d_in, const int* in_sizes, int n_in,
                              void* d_out, int out_size, void* d_ws, size_t ws_size,
                              hipStream_t stream) {
  const float* x  = (const float*)d_in[0];
  const float* Wq = (const float*)d_in[1];
  const float* Wk = (const float*)d_in[2];
  const float* Wv = (const float*)d_in[3];

  const size_t BT = 4 * 4096;
  half_t* Qw  = (half_t*)d_ws;            // [BT,64] f16
  half_t* Kw  = Qw + BT * 64;
  half_t* Vw  = Kw + BT * 64;
  half_t* W16 = Vw + BT * 64;             // [3,64,512] f16

  wcvt<<<96, 256, 0, stream>>>(Wq, Wk, Wv, W16);
  qkv_proj<<<128, 256, 0, stream>>>(x, W16, Qw, Kw, Vw);
  attn<<<4 * (4096 / 64), 128, 0, stream>>>(Qw, Kw, Vw, (float*)d_out);
}
